// SelfAttention_1554778161297
// MI455X (gfx1250) — matrix operand order repaired and checked by reading, not compiled, not run on hardware
//
#include <hip/hip_runtime.h>
#include <hip/hip_fp16.h>

// ---------------------------------------------------------------------------
// SAGAN self-attention for MI455X (gfx1250, wave32, WMMA + async LDS DMA).
//   B=4, C=512, D=64, N=64*64=4096.
// Pipeline:
//   proj_gemm<true>  : Q = Wf*x + bf  -> f16 [B, N, D]   (transposed store)
//   proj_gemm<true>  : K = Wg*x + bg  -> f16 [B, N, D]
//   proj_gemm<false> : V = Wh*x + bh  -> f16 [B, C, N]
//   attn_kernel      : flash-style softmax(Q^T K) * V^T, out = gamma*O + x
//                      K tiles double-buffered in LDS via async global->LDS.
// Workspace: 20 MB f16 (Q 2MB + K 2MB + V 16MB).
// ---------------------------------------------------------------------------

typedef __attribute__((ext_vector_type(16))) _Float16 v16h;
typedef __attribute__((ext_vector_type(8)))  _Float16 v8h;
typedef __attribute__((ext_vector_type(8)))  float    v8f;
typedef __attribute__((ext_vector_type(4)))  float    v4f;
typedef int b128v __attribute__((vector_size(16)));   // 16-byte unit for async DMA

static constexpr int Bc = 4;
static constexpr int Cc = 512;
static constexpr int Dd = 64;     // C/8
static constexpr int Nn = 4096;   // 64*64

#define WMMA_F32_F16(a, b, c) \
  __builtin_amdgcn_wmma_f32_16x16x32_f16(false, (a), false, (b), (short)0, (c), false, false)

// --- Async global->LDS copy (16B per lane), tracked by ASYNCcnt -------------
__device__ __forceinline__ void async_copy_b128(const _Float16* g, _Float16* l) {
#if __has_builtin(__builtin_amdgcn_global_load_async_to_lds_b128)
  __builtin_amdgcn_global_load_async_to_lds_b128(
      (__attribute__((address_space(1))) b128v*)g,
      (__attribute__((address_space(3))) b128v*)l, 0, 0);
#else
  unsigned loff = (unsigned)(uintptr_t)(__attribute__((address_space(3))) void*)l;
  asm volatile("global_load_async_to_lds_b128 %0, %1, off"
               :: "v"(loff), "v"(g) : "memory");
#endif
}

__device__ __forceinline__ void wait_async0() {
#if __has_builtin(__builtin_amdgcn_s_wait_asynccnt)
  __builtin_amdgcn_s_wait_asynccnt(0);
#else
  asm volatile("s_wait_asynccnt 0x0" ::: "memory");
#endif
}

// A fragment: 16x32 f16 tile, row-major source with leading dim `ld` (f16 elems).
// Per ISA 16-bit A layout: lane holds row (lane&15); elems 0..7 = cols kq..kq+7,
// elems 8..15 = cols kq+16..kq+23, kq = 8*(lane>>4).
__device__ __forceinline__ v16h load_fragA(const _Float16* base, int ld, int lane) {
  const int r  = lane & 15;
  const int kq = (lane >> 4) << 3;
  const _Float16* p = base + (size_t)r * ld + kq;
  v8h lo = *(const v8h*)(p);
  v8h hi = *(const v8h*)(p + 16);
  v16h out;
#pragma unroll
  for (int t = 0; t < 8; ++t) { out[t] = lo[t]; out[t + 8] = hi[t]; }
  return out;
}

// B fragment: 32x16 f16 tile; source holds B^T row-major (rows = n, cols = k),
// leading dim `ld`. Per ISA B layout: lane holds col n=(lane&15);
// elems 0..7 = k kq..kq+7, elems 8..15 = k kq+16..kq+23, kq = 8*(lane>>4).
__device__ __forceinline__ v16h load_fragB(const _Float16* baseT, int ld, int lane) {
  const int n = lane & 15;
  const int k = (lane >> 4) << 3;
  const _Float16* p = baseT + (size_t)n * ld + k;
  v8h lo = *(const v8h*)(p);
  v8h hi = *(const v8h*)(p + 16);
  v16h out;
#pragma unroll
  for (int t = 0; t < 8; ++t) { out[t] = lo[t]; out[t + 8] = hi[t]; }
  return out;
}

// ---------------------------------------------------------------------------
// Projection GEMM: Y[M,N] = W[M,C] * X[C,N] + bias, output f16.
// TRANSPOSE_OUT=1: store Y^T as [B, N, M] (for Q, K). Else [B, M, N] (for V).
// Block: 256 threads = 8 waves, block tile 64(m) x 32(n), waves 4x2.
// ---------------------------------------------------------------------------
template <bool TRANSPOSE_OUT>
__global__ __launch_bounds__(256) void proj_gemm_kernel(
    const float* __restrict__ x,    // [B, C, N]
    const float* __restrict__ Wm,   // [M, C]
    const float* __restrict__ bias, // [M]
    _Float16* __restrict__ outp,    // [B,M,N] or [B,N,M]
    int M) {
  constexpr int LDW = 40;  // padded f16 leading dims (multiple of 8 for 16B align)
  constexpr int LDX = 40;
  __shared__ _Float16 Ws[64 * LDW];  // W tile  [64 m][32 k]
  __shared__ _Float16 Xs[32 * LDX];  // X^T tile[32 n][32 k]

  const int tid  = threadIdx.x;
  const int lane = tid & 31;
  const int wave = tid >> 5;
  const int wm   = wave >> 1;  // 0..3
  const int wn   = wave & 1;   // 0..1
  const int n0   = blockIdx.x * 32;
  const int m0   = blockIdx.y * 64;
  const int b    = blockIdx.z;

  const float* xb = x + (size_t)b * Cc * Nn;

  v8f acc = {};
#pragma unroll 1
  for (int k0 = 0; k0 < Cc; k0 += 32) {
    __syncthreads();
    // Stage W tile [64 x 32] -> f16, row-major.
    {
      const int row = tid >> 2;         // 0..63
      const int col = (tid & 3) << 3;   // 0,8,16,24
      const float* wp = Wm + (size_t)(m0 + row) * Cc + k0 + col;
      v4f w0 = *(const v4f*)(wp);
      v4f w1 = *(const v4f*)(wp + 4);
      v8h h;
#pragma unroll
      for (int t = 0; t < 4; ++t) { h[t] = (_Float16)w0[t]; h[t + 4] = (_Float16)w1[t]; }
      *(v8h*)(&Ws[row * LDW + col]) = h;
    }
    // Stage X tile [32 k x 32 n] transposed -> Xs[n][k], f16.
    {
      const int krow = tid >> 3;        // 0..31
      const int ncol = (tid & 7) << 2;  // 0,4,...,28
      const float* xp = xb + (size_t)(k0 + krow) * Nn + n0 + ncol;
      v4f xv = *(const v4f*)(xp);
#pragma unroll
      for (int t = 0; t < 4; ++t) Xs[(ncol + t) * LDX + krow] = (_Float16)xv[t];
    }
    __syncthreads();
    v16h a  = load_fragA(&Ws[(wm * 16) * LDW], LDW, lane);
    v16h bf = load_fragB(&Xs[(wn * 16) * LDX], LDX, lane);
    acc = WMMA_F32_F16(a, bf, acc);
  }

  // Epilogue. C/D layout: value (r, lane) -> m_local = r + 8*(lane>>4), n_local = lane&15.
  const int hi = lane >> 4;
  const int nl = lane & 15;
  const int n_glob = n0 + wn * 16 + nl;
  const int mbase  = m0 + wm * 16 + hi * 8;
  if (TRANSPOSE_OUT) {
    v8h hv;
#pragma unroll
    for (int r = 0; r < 8; ++r) hv[r] = (_Float16)(acc[r] + bias[mbase + r]);
    *(v8h*)(outp + ((size_t)b * Nn + n_glob) * M + mbase) = hv;  // contiguous in m
  } else {
#pragma unroll
    for (int r = 0; r < 8; ++r) {
      const int m = mbase + r;
      outp[((size_t)b * M + m) * Nn + n_glob] = (_Float16)(acc[r] + bias[m]);
    }
  }
}

// ---------------------------------------------------------------------------
// Flash attention: one block per (batch, 16-query tile). 8 waves.
// All waves: prefetch next K tile (8KB contiguous) into LDS via async DMA.
// Wave 0: S = Q^T K from LDS (WMMA), online softmax (32 lanes), P -> LDS.
// All waves: O[c-range, 16] += V * P^T (WMMA), rescaled by alpha each step.
// ---------------------------------------------------------------------------
__global__ __launch_bounds__(256) void attn_kernel(
    const _Float16* __restrict__ Q,  // [B, N, D]
    const _Float16* __restrict__ K,  // [B, N, D]
    const _Float16* __restrict__ V,  // [B, C, N]
    const float* __restrict__ x,     // [B, C, N]
    const float* __restrict__ gamma, // [1]
    float* __restrict__ outp) {      // [B, C, N]
  __shared__ _Float16 Kbuf[2][64 * Dd]; // double-buffered K tile: 64 keys x 64 d
  __shared__ _Float16 Pbuf[16 * 64];    // probabilities, rows = i, cols = j
  __shared__ float    Sbuf[16 * 64];    // raw scores
  __shared__ float    alpha_s[16];
  __shared__ float    linv_s[16];

  const int tid  = threadIdx.x;
  const int lane = tid & 31;
  const int wave = tid >> 5;
  const int i0   = blockIdx.x * 16;
  const int b    = blockIdx.y;

  const _Float16* Qb = Q + (size_t)b * Nn * Dd;
  const _Float16* Kb = K + (size_t)b * Nn * Dd;
  const _Float16* Vb = V + (size_t)b * Cc * Nn;

  // Per-wave output accumulators: channels [wave*64, wave*64+64), 4 tiles of 16.
  v8f acc[4] = {{}, {}, {}, {}};

  // Prefetch first K tile (j = 0..63): contiguous 8KB, 2 x 16B per thread.
  async_copy_b128(Kb + tid * 8,        &Kbuf[0][0] + tid * 8);
  async_copy_b128(Kb + 2048 + tid * 8, &Kbuf[0][0] + 2048 + tid * 8);

  // Wave 0 state: Q fragments (rows i0..i0+15, d split 0..31 / 32..63) + softmax stats.
  v16h qa0 = {}, qa1 = {};
  float m_run = -1e30f, l_run = 0.f;
  if (wave == 0) {
    qa0 = load_fragA(Qb + (size_t)i0 * Dd, Dd, lane);
    qa1 = load_fragA(Qb + (size_t)i0 * Dd + 32, Dd, lane);
  }

  wait_async0();
  __syncthreads();

  int cur = 0;
#pragma unroll 1
  for (int j0 = 0; j0 < Nn; j0 += 64) {
    // Prefetch next K tile into the other buffer (overlaps with compute below).
    if (j0 + 64 < Nn) {
      const _Float16* src = Kb + (size_t)(j0 + 64) * Dd;
      _Float16* dst = &Kbuf[cur ^ 1][0];
      async_copy_b128(src + tid * 8,        dst + tid * 8);
      async_copy_b128(src + 2048 + tid * 8, dst + 2048 + tid * 8);
    }

    if (wave == 0) {
      // S tiles: 4 x (16i x 16j), contraction over D=64 via two WMMAs each.
      const _Float16* Kt = &Kbuf[cur][0];
#pragma unroll
      for (int jt = 0; jt < 4; ++jt) {
        v8f s = {};
        v16h kb0 = load_fragB(Kt + (jt * 16) * Dd, Dd, lane);
        s = WMMA_F32_F16(qa0, kb0, s);
        v16h kb1 = load_fragB(Kt + (jt * 16) * Dd + 32, Dd, lane);
        s = WMMA_F32_F16(qa1, kb1, s);
        const int hh = lane >> 4;
        const int jl = jt * 16 + (lane & 15);
#pragma unroll
        for (int r = 0; r < 8; ++r) Sbuf[(r + 8 * hh) * 64 + jl] = s[r];
      }
      // Online softmax: all 32 lanes. lane -> (row = lane&15, half = lane>>4);
      // each lane scans 32 scores, halves combined via shfl_xor(16).
      {
        const int row  = lane & 15;
        const int half = lane >> 4;
        const float* srow = &Sbuf[row * 64 + half * 32];
        float mloc = m_run;
        for (int j = 0; j < 32; ++j) mloc = fmaxf(mloc, srow[j]);
        const float mx = fmaxf(mloc, __shfl_xor(mloc, 16, 32));
        const float al = __expf(m_run - mx);
        float ls = 0.f;
        _Float16* prow = &Pbuf[row * 64 + half * 32];
        for (int j = 0; j < 32; ++j) {
          const float p = __expf(srow[j] - mx);
          prow[j] = (_Float16)p;
          ls += p;
        }
        ls += __shfl_xor(ls, 16, 32);
        l_run = al * l_run + ls;
        m_run = mx;
        if (half == 0) alpha_s[row] = al;
      }
    }
    __syncthreads();

    // All waves: rescale accumulators, then O += V[c, j0:j0+64] * P^T.
    const float al = alpha_s[lane & 15];
    v16h pb0 = load_fragB(Pbuf, 64, lane);        // j chunk 0..31
    v16h pb1 = load_fragB(Pbuf + 32, 64, lane);   // j chunk 32..63
    const int c0 = wave * 64;
#pragma unroll
    for (int ct = 0; ct < 4; ++ct) {
#pragma unroll
      for (int r = 0; r < 8; ++r) acc[ct][r] *= al;
      v16h va0 = load_fragA(Vb + (size_t)(c0 + ct * 16) * Nn + j0, Nn, lane);
      acc[ct] = WMMA_F32_F16(va0, pb0, acc[ct]);
      v16h va1 = load_fragA(Vb + (size_t)(c0 + ct * 16) * Nn + j0 + 32, Nn, lane);
      acc[ct] = WMMA_F32_F16(va1, pb1, acc[ct]);
    }

    wait_async0();    // next K tile landed in LDS
    __syncthreads();  // ... and Pbuf consumed by everyone
    cur ^= 1;
  }

  if (wave == 0 && lane < 16) linv_s[lane] = 1.0f / l_run;
  __syncthreads();

  // Epilogue: out = gamma * (O / l) + x. Value (ct, r, lane) ->
  //   c = wave*64 + ct*16 + 8*(lane>>4) + r, i = i0 + (lane&15).
  const float g  = gamma[0];
  const float li = linv_s[lane & 15];
  const int   ig = i0 + (lane & 15);
  const int   hi = lane >> 4;
#pragma unroll
  for (int ct = 0; ct < 4; ++ct) {
#pragma unroll
    for (int r = 0; r < 8; ++r) {
      const int c = wave * 64 + ct * 16 + hi * 8 + r;
      const size_t idx = ((size_t)b * Cc + c) * Nn + ig;
      outp[idx] = g * (acc[ct][r] * li) + x[idx];
    }
  }
}

// ---------------------------------------------------------------------------
extern "C" void kernel_launch(void* const* d_in, const int* in_sizes, int n_in,
                              void* d_out, int out_size, void* d_ws, size_t ws_size,
                              hipStream_t stream) {
  (void)in_sizes; (void)n_in; (void)out_size; (void)ws_size;

  const float* x     = (const float*)d_in[0];
  const float* Wf    = (const float*)d_in[1];
  const float* bfp   = (const float*)d_in[2];
  const float* Wg    = (const float*)d_in[3];
  const float* bgp   = (const float*)d_in[4];
  const float* Wh    = (const float*)d_in[5];
  const float* bhp   = (const float*)d_in[6];
  const float* gamma = (const float*)d_in[7];
  float* outp = (float*)d_out;

  _Float16* q_ws = (_Float16*)d_ws;                         // [B,N,D]  2 MB
  _Float16* k_ws = q_ws + (size_t)Bc * Nn * Dd;             // [B,N,D]  2 MB
  _Float16* v_ws = k_ws + (size_t)Bc * Nn * Dd;             // [B,C,N] 16 MB

  dim3 blk(256);
  dim3 gQK(Nn / 32, Dd / 64, Bc);
  proj_gemm_kernel<true><<<gQK, blk, 0, stream>>>(x, Wf, bfp, q_ws, Dd);
  proj_gemm_kernel<true><<<gQK, blk, 0, stream>>>(x, Wg, bgp, k_ws, Dd);
  dim3 gV(Nn / 32, Cc / 64, Bc);
  proj_gemm_kernel<false><<<gV, blk, 0, stream>>>(x, Wh, bhp, v_ws, Cc);
  dim3 gA(Nn / 16, Bc);
  attn_kernel<<<gA, blk, 0, stream>>>(q_ws, k_ws, v_ws, x, gamma, outp);
}
